// InvariantPointAttention_13838384628001
// MI455X (gfx1250) — matrix operand order repaired and checked by reading, not compiled, not run on hardware
//
#include <hip/hip_runtime.h>
#include <hip/hip_bf16.h>
#include <math.h>

// ---------------- problem constants ----------------
#define NN    512
#define CS    384
#define CZ    128
#define CH    16
#define NH    12
#define PQ    4
#define PV    8
#define CATW  2112           // H*(C_Z + C_H + 4*PV) = 12*176
#define INF_  100000.0f
#define EPS_  1e-8f

// cat layout offsets
#define OFF_O    0            // H*CH   = 192
#define OFF_X    192          // H*PV   = 96
#define OFF_Y    288
#define OFF_Z    384
#define OFF_NRM  480
#define OFF_PAIR 576          // H*CZ   = 1536  -> 2112 total

typedef __bf16 bf16;
typedef __attribute__((ext_vector_type(16))) __bf16 v16bf;
typedef __attribute__((ext_vector_type(8)))  __bf16 v8bf;
typedef __attribute__((ext_vector_type(4)))  __bf16 v4bf;
typedef __attribute__((ext_vector_type(8)))  float  v8f;
typedef __attribute__((ext_vector_type(4)))  float  v4f;
typedef __attribute__((ext_vector_type(4)))  int    v4i;

// ---- WMMA lane-layout helpers (wave32, 16x16x32 bf16) ----
// A (16x32, 16-bit): lane L -> M = L&15 ; element e -> K = (L>>4)*8 + e + (e>=8 ? 8 : 0)
// B (32x16, 16-bit): lane L -> N = L&15 ; element e -> K = (L>>4)*8 + e + (e>=8 ? 8 : 0)
// C/D (16x16 f32):   lane L, vgpr r -> M = r + 8*(L>>4), N = L&15
__device__ __forceinline__ int kA(int half, int e) { return half * 8 + e + ((e >= 8) ? 8 : 0); }
__device__ __forceinline__ int kB(int half, int e) { return half * 8 + e + ((e >= 8) ? 8 : 0); }

__device__ __forceinline__ v8bf cvt8(v4f a, v4f b) {
  v8bf r;
  r[0] = (bf16)a[0]; r[1] = (bf16)a[1]; r[2] = (bf16)a[2]; r[3] = (bf16)a[3];
  r[4] = (bf16)b[0]; r[5] = (bf16)b[1]; r[6] = (bf16)b[2]; r[7] = (bf16)b[3];
  return r;
}
__device__ __forceinline__ v16bf cat16(v8bf lo, v8bf hi) {
  return __builtin_shufflevector(lo, hi, 0,1,2,3,4,5,6,7,8,9,10,11,12,13,14,15);
}
__device__ __forceinline__ v8bf zero8() {
  v8bf z;
#pragma unroll
  for (int e = 0; e < 8; ++e) z[e] = (bf16)0.0f;
  return z;
}
// A operand, fp32 source with contiguous K; p = row base + k0; two 8-elem groups (+0, +16)
__device__ __forceinline__ v16bf loadA_f32(const float* p, int half, float s) {
  const v4f* q = (const v4f*)(p + half * 8);
  v4f a0 = q[0] * s, a1 = q[1] * s, a2 = q[4] * s, a3 = q[5] * s;
  return cat16(cvt8(a0, a1), cvt8(a2, a3));
}
// A operand, bf16 source with contiguous K (attention probabilities)
__device__ __forceinline__ v16bf loadA_bf(const bf16* p, int half) {
  const v8bf* q = (const v8bf*)(p + half * 8);
  return cat16(q[0], q[2]);
}
// B operand where K is contiguous in memory; p = k0 element of this lane's column row
__device__ __forceinline__ v16bf loadB_kcontig(const float* p, int half, float s) {
  const v4f* q = (const v4f*)(p + half * 16);
  return cat16(cvt8(q[0] * s, q[1] * s), cvt8(q[2] * s, q[3] * s));
}

// ---- CDNA5 async global->LDS copy (ASYNCcnt path), guarded fallbacks ----
__device__ __forceinline__ void async_cp_b128(const float* g, float* l) {
#if __has_builtin(__builtin_amdgcn_global_load_async_to_lds_b128)
  __builtin_amdgcn_global_load_async_to_lds_b128(
      (__attribute__((address_space(1))) v4i*)g,
      (__attribute__((address_space(3))) v4i*)l, 0, 0);
#else
  *(v4f*)l = *(const v4f*)g;
#endif
}
__device__ __forceinline__ void wait_async0() {
#if __has_builtin(__builtin_amdgcn_s_wait_asynccnt)
  __builtin_amdgcn_s_wait_asynccnt(0);
#else
  asm volatile("s_wait_asynccnt 0x0" ::: "memory");
#endif
}

// =====================================================================
// 1) Generic projection GEMM: out[M x ncols] = A[M x Kdim] @ W + bias
// =====================================================================
__global__ __launch_bounds__(32)
void proj_gemm(const float* __restrict__ A, const float* __restrict__ W,
               const float* __restrict__ bias, float* __restrict__ out,
               int Kdim, int ncols) {
  const int ntc  = ncols >> 4;
  const int row0 = (blockIdx.x / ntc) * 16;
  const int col0 = (blockIdx.x % ntc) * 16;
  const int lane = threadIdx.x, half = lane >> 4, lm = lane & 15;

  v8f acc = {};
  for (int k0 = 0; k0 < Kdim; k0 += 32) {
    v16bf a = loadA_f32(A + (row0 + lm) * Kdim + k0, half, 1.0f);
    v16bf b;
#pragma unroll
    for (int e = 0; e < 16; ++e)
      b[e] = (bf16)W[(k0 + kB(half, e)) * ncols + col0 + lm];
    acc = __builtin_amdgcn_wmma_f32_16x16x32_bf16(false, a, false, b,
                                                  (short)0, acc, false, false);
  }
#pragma unroll
  for (int r = 0; r < 8; ++r) {
    int m = row0 + r + 8 * half, n = col0 + lm;
    out[m * ncols + n] = acc[r] + bias[n];
  }
}

// =====================================================================
// 2) Frame transforms: pl -> rot @ pl + trans (q_pts / split kv_pts)
// =====================================================================
__global__ void qpts_transform(const float* __restrict__ lin, const float* __restrict__ rot,
                               const float* __restrict__ trans, float* __restrict__ pts) {
  int idx = blockIdx.x * blockDim.x + threadIdx.x;     // n*H*PQ + h*PQ + p
  if (idx >= NN * NH * PQ) return;
  int n = idx / (NH * PQ), hp = idx % (NH * PQ);
  float x = lin[n * (3 * NH * PQ) + 0 * (NH * PQ) + hp];
  float y = lin[n * (3 * NH * PQ) + 1 * (NH * PQ) + hp];
  float z = lin[n * (3 * NH * PQ) + 2 * (NH * PQ) + hp];
  const float* R = rot + n * 9;
  pts[idx * 3 + 0] = R[0] * x + R[1] * y + R[2] * z + trans[n * 3 + 0];
  pts[idx * 3 + 1] = R[3] * x + R[4] * y + R[5] * z + trans[n * 3 + 1];
  pts[idx * 3 + 2] = R[6] * x + R[7] * y + R[8] * z + trans[n * 3 + 2];
}

__global__ void kvpts_transform(const float* __restrict__ lin, const float* __restrict__ rot,
                                const float* __restrict__ trans,
                                float* __restrict__ k_pts, float* __restrict__ v_pts) {
  const int P = PQ + PV;
  int idx = blockIdx.x * blockDim.x + threadIdx.x;     // n*H*P + h*P + p
  if (idx >= NN * NH * P) return;
  int n = idx / (NH * P), hp = idx % (NH * P);
  int h = hp / P, p = hp % P;
  float x = lin[n * (3 * NH * P) + 0 * (NH * P) + hp];
  float y = lin[n * (3 * NH * P) + 1 * (NH * P) + hp];
  float z = lin[n * (3 * NH * P) + 2 * (NH * P) + hp];
  const float* R = rot + n * 9;
  float gx = R[0] * x + R[1] * y + R[2] * z + trans[n * 3 + 0];
  float gy = R[3] * x + R[4] * y + R[5] * z + trans[n * 3 + 1];
  float gz = R[6] * x + R[7] * y + R[8] * z + trans[n * 3 + 2];
  float* dst = (p < PQ) ? (k_pts + ((n * NH + h) * PQ + p) * 3)
                        : (v_pts + ((n * NH + h) * PV + (p - PQ)) * 3);
  dst[0] = gx; dst[1] = gy; dst[2] = gz;
}

// =====================================================================
// 3) attn[h,i,j] = sqrt(1/3) * (z[i,j,:] @ Wb[:,h] + bb[h])
//    rows = 16 flattened (i,j) pairs (fully coalesced b128 on z), cols = heads
// =====================================================================
__global__ __launch_bounds__(32)
void bias_gemm(const float* __restrict__ z, const float* __restrict__ Wb,
               const float* __restrict__ bb, float* __restrict__ attn) {
  const float SQ13 = 0.57735026919f;
  const int t0 = blockIdx.x * 16;                       // flattened i*N+j base
  const int lane = threadIdx.x, half = lane >> 4, lm = lane & 15;

  v8f acc = {};
  for (int k0 = 0; k0 < CZ; k0 += 32) {
    v16bf a = loadA_f32(z + (size_t)(t0 + lm) * CZ + k0, half, SQ13);
    v16bf b;
#pragma unroll
    for (int e = 0; e < 16; ++e)
      b[e] = (lm < NH) ? (bf16)Wb[(k0 + kB(half, e)) * NH + lm] : (bf16)0.0f;
    acc = __builtin_amdgcn_wmma_f32_16x16x32_bf16(false, a, false, b,
                                                  (short)0, acc, false, false);
  }
  if (lm < NH) {
    const float bbv = SQ13 * bb[lm];
    float* dst = attn + (size_t)lm * NN * NN + t0 + 8 * half;   // 8 consecutive t
    v4f s0 = {acc[0] + bbv, acc[1] + bbv, acc[2] + bbv, acc[3] + bbv};
    v4f s1 = {acc[4] + bbv, acc[5] + bbv, acc[6] + bbv, acc[7] + bbv};
    *(v4f*)(dst)     = s0;
    *(v4f*)(dst + 4) = s1;
  }
}

// =====================================================================
// 4) attn[h,i,j] += scale * q[i,h,:] . k[j,h,:]    (K=16 padded to 32)
//    C operand loaded from attn (bias already there) -> free accumulate
// =====================================================================
__global__ __launch_bounds__(32)
void qk_gemm(const float* __restrict__ q_lin, const float* __restrict__ kv_lin,
             float* __restrict__ attn) {
  const float QKS = 0.14433756729f;                     // sqrt(1/(3*16))
  int b = blockIdx.x;
  const int h = b / (32 * 32); b %= (32 * 32);
  const int i0 = (b / 32) * 16, j0 = (b % 32) * 16;
  const int lane = threadIdx.x, half = lane >> 4, lm = lane & 15;

  // A: valid K = half*8 .. half*8+7 (contiguous 8 floats), rest zero padding
  v16bf a;
  {
    const v4f* qp = (const v4f*)(q_lin + (i0 + lm) * (NH * CH) + h * CH + half * 8);
    a = cat16(cvt8(qp[0] * QKS, qp[1] * QKS), zero8());
  }
  // B: valid K = half*8 .. half*8+7 (contiguous 8 floats), rest zero padding
  v16bf bm;
  {
    const v4f* kp = (const v4f*)(kv_lin + (j0 + lm) * (2 * NH * CH) + h * 2 * CH + half * 8);
    bm = cat16(cvt8(kp[0], kp[1]), zero8());
  }
  v8f c;
#pragma unroll
  for (int r = 0; r < 8; ++r)
    c[r] = attn[(size_t)h * NN * NN + (size_t)(i0 + r + 8 * half) * NN + j0 + lm];
  c = __builtin_amdgcn_wmma_f32_16x16x32_bf16(false, a, false, bm,
                                              (short)0, c, false, false);
#pragma unroll
  for (int r = 0; r < 8; ++r)
    attn[(size_t)h * NN * NN + (size_t)(i0 + r + 8 * half) * NN + j0 + lm] = c[r];
}

// =====================================================================
// 5) attn[h,i,j] += -0.5*hw(h)*sum_p |q_pts[i,h,p]-k_pts[j,h,p]|^2 + mask
// =====================================================================
__global__ __launch_bounds__(256)
void pt_mask(const float* __restrict__ q_pts, const float* __restrict__ k_pts,
             const float* __restrict__ mask, const float* __restrict__ head_weights,
             float* __restrict__ attn) {
  const int h = blockIdx.x / NN, i = blockIdx.x % NN;
  const float hw = log1pf(__expf(head_weights[h])) * sqrtf(1.0f / (3.0f * (PQ * 9.0f / 2.0f)));
  const v4f* qp = (const v4f*)(q_pts + (size_t)i * NH * PQ * 3 + h * PQ * 3);
  v4f q0 = qp[0], q1 = qp[1], q2 = qp[2];               // 12 floats = PQ*3
  const float mi = mask[i];
  float* row = attn + (size_t)h * NN * NN + (size_t)i * NN;
  for (int j = threadIdx.x; j < NN; j += blockDim.x) {
    const v4f* kp = (const v4f*)(k_pts + (size_t)j * NH * PQ * 3 + h * PQ * 3);
    v4f d0 = q0 - kp[0], d1 = q1 - kp[1], d2v = q2 - kp[2];
    float d2 = d0[0]*d0[0] + d0[1]*d0[1] + d0[2]*d0[2] + d0[3]*d0[3]
             + d1[0]*d1[0] + d1[1]*d1[1] + d1[2]*d1[2] + d1[3]*d1[3]
             + d2v[0]*d2v[0] + d2v[1]*d2v[1] + d2v[2]*d2v[2] + d2v[3]*d2v[3];
    row[j] += -0.5f * hw * d2 + INF_ * (mi * mask[j] - 1.0f);
  }
}

// =====================================================================
// 6) row softmax over j; emit bf16 copy of probabilities (v4 loads/stores)
// =====================================================================
__global__ __launch_bounds__(32)
void softmax_rows(float* __restrict__ attn, bf16* __restrict__ attn_bf) {
  const size_t row = blockIdx.x;                        // h*N + i
  float* p = attn + row * NN;
  const int t = threadIdx.x;
  float m = -INF_;
  for (int jb = t * 4; jb < NN; jb += 128) {
    v4f v = *(const v4f*)(p + jb);
    m = fmaxf(fmaxf(fmaxf(m, v[0]), fmaxf(v[1], v[2])), v[3]);
  }
#pragma unroll
  for (int off = 16; off > 0; off >>= 1) m = fmaxf(m, __shfl_xor(m, off, 32));
  float s = 0.0f;
  for (int jb = t * 4; jb < NN; jb += 128) {
    v4f v = *(const v4f*)(p + jb);
    s += __expf(v[0] - m) + __expf(v[1] - m) + __expf(v[2] - m) + __expf(v[3] - m);
  }
#pragma unroll
  for (int off = 16; off > 0; off >>= 1) s += __shfl_xor(s, off, 32);
  const float inv = 1.0f / s;
  for (int jb = t * 4; jb < NN; jb += 128) {
    v4f v = *(const v4f*)(p + jb);
    v4f e = {__expf(v[0] - m) * inv, __expf(v[1] - m) * inv,
             __expf(v[2] - m) * inv, __expf(v[3] - m) * inv};
    *(v4f*)(p + jb) = e;
    v4bf eb; eb[0] = (bf16)e[0]; eb[1] = (bf16)e[1]; eb[2] = (bf16)e[2]; eb[3] = (bf16)e[3];
    *(v4bf*)(attn_bf + row * NN + jb) = eb;
  }
}

// =====================================================================
// 7) o[i, h*16+c] = sum_j a[h,i,j] v[j,h,c]  -> cat[:, OFF_O + h*16 + c]
// =====================================================================
__global__ __launch_bounds__(32)
void av_gemm(const bf16* __restrict__ attn_bf, const float* __restrict__ kv_lin,
             float* __restrict__ cat) {
  const int h = blockIdx.x / 32, i0 = (blockIdx.x % 32) * 16;
  const int lane = threadIdx.x, half = lane >> 4, lm = lane & 15;
  v8f acc = {};
  for (int k0 = 0; k0 < NN; k0 += 32) {
    v16bf a = loadA_bf(attn_bf + (size_t)h * NN * NN + (size_t)(i0 + lm) * NN + k0, half);
    v16bf b;
#pragma unroll
    for (int e = 0; e < 16; ++e) {
      int j = k0 + kB(half, e);
      b[e] = (bf16)kv_lin[j * (2 * NH * CH) + h * 2 * CH + CH + lm];   // v part
    }
    acc = __builtin_amdgcn_wmma_f32_16x16x32_bf16(false, a, false, b,
                                                  (short)0, acc, false, false);
  }
#pragma unroll
  for (int r = 0; r < 8; ++r)
    cat[(size_t)(i0 + r + 8 * half) * CATW + OFF_O + h * CH + lm] = acc[r];
}

// =====================================================================
// 8) o_pt_global[i,h,p,c] = sum_j a[h,i,j] v_pts[j,h,p,c]   (24 cols -> 2 tiles)
// =====================================================================
__global__ __launch_bounds__(32)
void apts_gemm(const bf16* __restrict__ attn_bf, const float* __restrict__ v_pts,
               float* __restrict__ o_ptg) {
  int b = blockIdx.x;
  const int ct = b & 1; b >>= 1;
  const int h = b / 32, i0 = (b % 32) * 16;
  const int lane = threadIdx.x, half = lane >> 4, lm = lane & 15;
  const int n = ct * 16 + lm;                            // p*3+c, valid < 24
  const bool valid = n < PV * 3;
  v8f acc = {};
  for (int k0 = 0; k0 < NN; k0 += 32) {
    v16bf a = loadA_bf(attn_bf + (size_t)h * NN * NN + (size_t)(i0 + lm) * NN + k0, half);
    v16bf bm;
#pragma unroll
    for (int e = 0; e < 16; ++e) {
      int j = k0 + kB(half, e);
      bm[e] = valid ? (bf16)v_pts[(j * NH + h) * PV * 3 + n] : (bf16)0.0f;
    }
    acc = __builtin_amdgcn_wmma_f32_16x16x32_bf16(false, a, false, bm,
                                                  (short)0, acc, false, false);
  }
  if (valid) {
#pragma unroll
    for (int r = 0; r < 8; ++r)
      o_ptg[((i0 + r + 8 * half) * NH + h) * PV * 3 + n] = acc[r];
  }
}

// =====================================================================
// 9) o_pair[i,h,c] = sum_j a[h,i,j] z[i,j,c]
//    One 256-thread block (8 waves) per i. The 32x128 fp32 z tile is staged
//    into LDS with async b128 copies (double-buffered, one barrier/step) so
//    the hot 134MB z pass is fully coalesced; LDS rows padded (132) for
//    conflict-free strided reads.
// =====================================================================
__global__ __launch_bounds__(256)
void az_gemm(const bf16* __restrict__ attn_bf, const float* __restrict__ z,
             float* __restrict__ cat) {
  __shared__ float zs[2][32][CZ + 4];
  const int i    = blockIdx.x;
  const int t    = threadIdx.x;
  const int lane = t & 31, half = lane >> 4, lm = lane & 15;
  const int c0   = (t >> 5) * 16;                        // wave -> column tile
  const float* zrow = z + (size_t)i * NN * CZ;

  // prologue: stage K-tile 0
#pragma unroll
  for (int q = 0; q < 4; ++q) {
    int f = t * 4 + q * 1024;                            // flat float idx in 32x128 tile
    int r = f >> 7, c = f & 127;
    async_cp_b128(zrow + (size_t)r * CZ + c, &zs[0][r][c]);
  }

  v8f acc = {};
  for (int kt = 0; kt < NN / 32; ++kt) {
    wait_async0();
    __syncthreads();                                     // tile kt resident for all waves
    if (kt + 1 < NN / 32) {                              // prefetch kt+1 (other buffer)
      const float* g = zrow + (size_t)(kt + 1) * 32 * CZ;
#pragma unroll
      for (int q = 0; q < 4; ++q) {
        int f = t * 4 + q * 1024;
        int r = f >> 7, c = f & 127;
        async_cp_b128(g + (size_t)r * CZ + c, &zs[(kt + 1) & 1][r][c]);
      }
    }
    const int buf = kt & 1;
    v16bf a;
    if (lm < NH) {
      a = loadA_bf(attn_bf + (size_t)lm * NN * NN + (size_t)i * NN + kt * 32, half);
    } else {
      a = cat16(zero8(), zero8());
    }
    v16bf b;
#pragma unroll
    for (int e = 0; e < 16; ++e)
      b[e] = (bf16)zs[buf][kB(half, e)][c0 + lm];
    acc = __builtin_amdgcn_wmma_f32_16x16x32_bf16(false, a, false, b,
                                                  (short)0, acc, false, false);
  }
#pragma unroll
  for (int r = 0; r < 8; ++r) {
    int h = r + 8 * half;
    if (h < NH)
      cat[(size_t)i * CATW + OFF_PAIR + h * CZ + c0 + lm] = acc[r];
  }
}

// =====================================================================
// 10) o_pt -> local frame (R^T (g - t)), norms, fill cat x/y/z/norm
// =====================================================================
__global__ void opt_finish(const float* __restrict__ o_ptg, const float* __restrict__ rot,
                           const float* __restrict__ trans, float* __restrict__ cat) {
  int idx = blockIdx.x * blockDim.x + threadIdx.x;       // n*H*PV + hp
  if (idx >= NN * NH * PV) return;
  int n = idx / (NH * PV), hp = idx % (NH * PV);
  const float* R = rot + n * 9;
  float gx = o_ptg[idx * 3 + 0] - trans[n * 3 + 0];
  float gy = o_ptg[idx * 3 + 1] - trans[n * 3 + 1];
  float gz = o_ptg[idx * 3 + 2] - trans[n * 3 + 2];
  float lx = R[0] * gx + R[3] * gy + R[6] * gz;
  float ly = R[1] * gx + R[4] * gy + R[7] * gz;
  float lz = R[2] * gx + R[5] * gy + R[8] * gz;
  float* c = cat + (size_t)n * CATW;
  c[OFF_X + hp] = lx;
  c[OFF_Y + hp] = ly;
  c[OFF_Z + hp] = lz;
  c[OFF_NRM + hp] = sqrtf(lx * lx + ly * ly + lz * lz + EPS_);
}

// =====================================================================
// 11) out = cat @ Wout + bout   (512 x 2112 x 384)
// =====================================================================
__global__ __launch_bounds__(32)
void out_gemm(const float* __restrict__ cat, const float* __restrict__ Wout,
              const float* __restrict__ bout, float* __restrict__ out) {
  const int row0 = (blockIdx.x / (CS / 16)) * 16, col0 = (blockIdx.x % (CS / 16)) * 16;
  const int lane = threadIdx.x, half = lane >> 4, lm = lane & 15;
  v8f acc = {};
  for (int k0 = 0; k0 < CATW; k0 += 32) {
    v16bf a = loadA_f32(cat + (size_t)(row0 + lm) * CATW + k0, half, 1.0f);
    v16bf b;
#pragma unroll
    for (int e = 0; e < 16; ++e)
      b[e] = (bf16)Wout[(size_t)(k0 + kB(half, e)) * CS + col0 + lm];
    acc = __builtin_amdgcn_wmma_f32_16x16x32_bf16(false, a, false, b,
                                                  (short)0, acc, false, false);
  }
#pragma unroll
  for (int r = 0; r < 8; ++r)
    out[(size_t)(row0 + r + 8 * half) * CS + col0 + lm] = acc[r] + bout[col0 + lm];
}

// =====================================================================
// launcher
// =====================================================================
extern "C" void kernel_launch(void* const* d_in, const int* in_sizes, int n_in,
                              void* d_out, int out_size, void* d_ws, size_t ws_size,
                              hipStream_t stream) {
  const float* s     = (const float*)d_in[0];
  const float* z     = (const float*)d_in[1];
  const float* rot   = (const float*)d_in[2];
  const float* trans = (const float*)d_in[3];
  const float* mask  = (const float*)d_in[4];
  const float* Wq    = (const float*)d_in[5];
  const float* bq    = (const float*)d_in[6];
  const float* Wkv   = (const float*)d_in[7];
  const float* bkv   = (const float*)d_in[8];
  const float* Wqp   = (const float*)d_in[9];
  const float* bqp   = (const float*)d_in[10];
  const float* Wkvp  = (const float*)d_in[11];
  const float* bkvp  = (const float*)d_in[12];
  const float* Wb    = (const float*)d_in[13];
  const float* bb    = (const float*)d_in[14];
  const float* hwts  = (const float*)d_in[15];
  const float* Wout  = (const float*)d_in[16];
  const float* bout  = (const float*)d_in[17];
  float* out = (float*)d_out;

  // ---- workspace layout (floats) ----
  float* ws = (float*)d_ws;
  float* q_lin   = ws;                 // N*192
  float* kv_lin  = q_lin  + NN * 192;  // N*384
  float* qp_lin  = kv_lin + NN * 384;  // N*144
  float* kvp_lin = qp_lin + NN * 144;  // N*432
  float* q_pts   = kvp_lin + NN * 432; // N*H*PQ*3
  float* k_pts   = q_pts  + NN * NH * PQ * 3;
  float* v_pts   = k_pts  + NN * NH * PQ * 3;
  float* attn    = v_pts  + NN * NH * PV * 3;              // H*N*N f32
  float* o_ptg   = attn   + (size_t)NH * NN * NN;          // N*H*PV*3
  float* cat     = o_ptg  + NN * NH * PV * 3;              // N*2112
  bf16*  attn_bf = (bf16*)(cat + (size_t)NN * CATW);       // H*N*N bf16

  // 1) projections (bf16 WMMA, fp32 accumulate)
  proj_gemm<<<dim3(32 * (192 / 16)), dim3(32), 0, stream>>>(s, Wq,   bq,   q_lin,   CS, 192);
  proj_gemm<<<dim3(32 * (384 / 16)), dim3(32), 0, stream>>>(s, Wkv,  bkv,  kv_lin,  CS, 384);
  proj_gemm<<<dim3(32 * (144 / 16)), dim3(32), 0, stream>>>(s, Wqp,  bqp,  qp_lin,  CS, 144);
  proj_gemm<<<dim3(32 * (432 / 16)), dim3(32), 0, stream>>>(s, Wkvp, bkvp, kvp_lin, CS, 432);

  // 2) frame transforms
  qpts_transform<<<dim3((NN * NH * PQ + 255) / 256), dim3(256), 0, stream>>>(qp_lin, rot, trans, q_pts);
  kvpts_transform<<<dim3((NN * NH * (PQ + PV) + 255) / 256), dim3(256), 0, stream>>>(kvp_lin, rot, trans, k_pts, v_pts);

  // 3) logits: bias term, then QK accumulate, then point term + mask
  bias_gemm<<<dim3(NN * NN / 16), dim3(32), 0, stream>>>(z, Wb, bb, attn);
  qk_gemm<<<dim3(NH * 32 * 32), dim3(32), 0, stream>>>(q_lin, kv_lin, attn);
  pt_mask<<<dim3(NH * NN), dim3(256), 0, stream>>>(q_pts, k_pts, mask, hwts, attn);

  // 4) softmax -> bf16 probs
  softmax_rows<<<dim3(NH * NN), dim3(32), 0, stream>>>(attn, attn_bf);

  // 5) attention-weighted reductions (all WMMA)
  av_gemm<<<dim3(NH * 32), dim3(32), 0, stream>>>(attn_bf, kv_lin, cat);
  apts_gemm<<<dim3(NH * 32 * 2), dim3(32), 0, stream>>>(attn_bf, v_pts, o_ptg);
  az_gemm<<<dim3(NN), dim3(256), 0, stream>>>(attn_bf, z, cat);

  // 6) local-frame points + norms into cat
  opt_finish<<<dim3((NN * NH * PV + 255) / 256), dim3(256), 0, stream>>>(o_ptg, rot, trans, cat);

  // 7) final projection
  out_gemm<<<dim3(32 * (CS / 16)), dim3(32), 0, stream>>>(cat, Wout, bout, out);
}
